// MSDeformAttnTransformerEncoderLayer_5566277615775
// MI455X (gfx1250) — matrix operand order repaired and checked by reading, not compiled, not run on hardware
//
#include <hip/hip_runtime.h>
#include <hip/hip_bf16.h>
#include <math.h>

// ---------------------------------------------------------------------------
// Problem constants (match reference)
// ---------------------------------------------------------------------------
#define BATCH      2
#define LEN_TOK    21760             // 128*128 + 64*64 + 32*32 + 16*16
#define M_TOK      (BATCH * LEN_TOK) // 43520
#define D_MODEL    256
#define D_FFN      1024
#define N_HEADS    8
#define N_LEVELS   4
#define N_POINTS   4
#define D_HEAD     32

typedef __attribute__((ext_vector_type(16))) _Float16 v16h;
typedef __attribute__((ext_vector_type(8)))  _Float16 v8h;
typedef __attribute__((ext_vector_type(8)))  float    v8f;
typedef __attribute__((ext_vector_type(4)))  float    f32x4;

// ---------------------------------------------------------------------------
// GEMM: C[M,N] = (A (+A2)) @ W[K,N] + bias, optional ReLU, optional row mask.
// fp32 in/out, f16 WMMA compute with fp32 accumulate.
//
// Block = 256 threads = 8 waves.  Block tile 64x64, K-step 32.
//   waves: (wave&3) -> M strip of 16, (wave>>2) -> 32-col group; each wave
//   owns a 16x32 strip = two 16x16 WMMA tiles sharing one A fragment.
// A tile (64x32) and W tile (32x64, stored transposed) staged in LDS as f16.
// Requires M%64==0, N%64==0, K%32==0 (true for all six GEMMs here).
// ---------------------------------------------------------------------------
template<bool HAS_A2>
__global__ __launch_bounds__(256)
void gemm_wmma_f16(const float* __restrict__ A,
                   const float* __restrict__ A2,   // added to A when HAS_A2
                   const float* __restrict__ W,
                   const float* __restrict__ bias,
                   float* __restrict__ C,
                   int M, int N, int K,
                   int do_relu,
                   const unsigned char* __restrict__ row_mask)
{
    __shared__ _Float16 sA[64][32];   // [m][k]  4 KB
    __shared__ _Float16 sB[64][32];   // [n][k]  4 KB (transposed W tile)

    const int tid   = threadIdx.x;
    const int lane  = tid & 31;
    const int wave  = tid >> 5;
    const int waveM = wave & 3;        // 0..3  (16-row strip)
    const int waveN = wave >> 2;       // 0..1  (32-col group)

    const int m0 = blockIdx.y * 64;    // block rows
    const int n0 = blockIdx.x * 64;    // block cols

    const int r15    = lane & 15;
    const int half   = lane >> 4;      // 0: lanes 0-15, 1: lanes 16-31
    const int half8  = half << 3;      // A-fragment K offset (upper half-wave)
    const int half16 = half << 4;      // unused; B fragment also uses half8

    // cooperative load indices
    const int rA = tid >> 2;           // 0..63   A row within tile
    const int cA = (tid & 3) << 3;     // 0,8,16,24  A col group (8 floats)
    const int kB = tid >> 3;           // 0..31   W row within tile
    const int nB = (tid & 7) << 3;     // 0..56   W col group (8 floats)

    v8f acc0 = {};
    v8f acc1 = {};

    for (int k0 = 0; k0 < K; k0 += 32) {
        // ---- stage A tile (f32 -> f16) --------------------------------
        {
            const size_t base = (size_t)(m0 + rA) * K + k0 + cA;
            f32x4 a0 = *(const f32x4*)(A + base);
            f32x4 a1 = *(const f32x4*)(A + base + 4);
            if (HAS_A2) {
                f32x4 p0 = *(const f32x4*)(A2 + base);
                f32x4 p1 = *(const f32x4*)(A2 + base + 4);
                a0 += p0; a1 += p1;
            }
            v8h h;
#pragma unroll
            for (int j = 0; j < 4; ++j) { h[j] = (_Float16)a0[j]; h[4 + j] = (_Float16)a1[j]; }
            *(v8h*)&sA[rA][cA] = h;
        }
        // ---- stage W tile transposed (f32 -> f16) ---------------------
        {
            const size_t base = (size_t)(k0 + kB) * N + n0 + nB;
            f32x4 b0 = *(const f32x4*)(W + base);
            f32x4 b1 = *(const f32x4*)(W + base + 4);
#pragma unroll
            for (int j = 0; j < 4; ++j) {
                sB[nB + j][kB]     = (_Float16)b0[j];
                sB[nB + 4 + j][kB] = (_Float16)b1[j];
            }
        }
        // prefetch next K-tile while this one is consumed
        if (k0 + 32 < K) {
            __builtin_prefetch(A + (size_t)(m0 + rA) * K + k0 + 32 + cA, 0, 1);
            __builtin_prefetch(W + (size_t)(k0 + 32 + kB) * N + n0 + nB, 0, 1);
        }
        __syncthreads();

        // ---- fragments from LDS ---------------------------------------
        v16h afrag, b0frag, b1frag;
        const int arow = waveM * 16 + r15;
        const int ncol0 = waveN * 32 + r15;
#pragma unroll
        for (int i = 0; i < 16; ++i) {
            // 16-bit A 16x32 layout: k = i + (i&8) + 8*(lane>=16)
            afrag[i]  = sA[arow][i + (i & 8) + half8];
            // 16-bit B 32x16 layout: k = i + (i&8) + 8*(lane>=16)
            b0frag[i] = sB[ncol0][i + (i & 8) + half8];
            b1frag[i] = sB[ncol0 + 16][i + (i & 8) + half8];
        }
        acc0 = __builtin_amdgcn_wmma_f32_16x16x32_f16(
            false, afrag, false, b0frag, (short)0, acc0, false, false);
        acc1 = __builtin_amdgcn_wmma_f32_16x16x32_f16(
            false, afrag, false, b1frag, (short)0, acc1, false, false);
        __syncthreads();
    }

    // ---- epilogue: C layout VGPR j -> row = base + j + 8*half ---------
    const int rowT = m0 + waveM * 16;
    const int colA = n0 + waveN * 32 + r15;
    const int colB = colA + 16;
    const float bvA = bias ? bias[colA] : 0.0f;
    const float bvB = bias ? bias[colB] : 0.0f;
#pragma unroll
    for (int j = 0; j < 8; ++j) {
        const int r = rowT + j + (half << 3);
        float va = acc0[j] + bvA;
        float vb = acc1[j] + bvB;
        if (do_relu) { va = fmaxf(va, 0.0f); vb = fmaxf(vb, 0.0f); }
        if (row_mask && row_mask[r]) { va = 0.0f; vb = 0.0f; }
        C[(size_t)r * N + colA] = va;
        C[(size_t)r * N + colB] = vb;
    }
}

// ---------------------------------------------------------------------------
// Multi-scale deformable sampling.
// One wave per (token, head); 32 lanes = 32 channels of D_HEAD.
// ---------------------------------------------------------------------------
__device__ __forceinline__ float ms_corner(const float* __restrict__ value,
                                           int b, int start, int Hl, int Wl,
                                           int xi, int yi, float w,
                                           int head, int lane)
{
    const bool valid = (xi >= 0) && (xi < Wl) && (yi >= 0) && (yi < Hl);
    const int xc = min(max(xi, 0), Wl - 1);
    const int yc = min(max(yi, 0), Hl - 1);
    const size_t idx = ((size_t)b * LEN_TOK + start + (size_t)yc * Wl + xc) * D_MODEL
                     + head * D_HEAD + lane;
    const float v = value[idx];
    return v * (valid ? w : 0.0f);
}

__global__ __launch_bounds__(256)
void msdeform_sample(const float* __restrict__ value,
                     const float* __restrict__ off_raw,
                     const float* __restrict__ attn_raw,
                     const float* __restrict__ ref,
                     float* __restrict__ out)
{
    const int lane = threadIdx.x & 31;
    const int wave = threadIdx.x >> 5;
    const int g    = blockIdx.x * 8 + wave;   // global wave index
    const int t    = g >> 3;                  // token index in [0, M_TOK)
    const int h    = g & 7;                   // head
    const int b    = t / LEN_TOK;

    const int Hs[N_LEVELS]     = {128, 64, 32, 16};
    const int Ws[N_LEVELS]     = {128, 64, 32, 16};
    const int starts[N_LEVELS] = {0, 16384, 20480, 21504};

    // softmax over the 16 (level,point) logits -- uniform across lanes
    const size_t at_base = (size_t)t * (N_HEADS * 16) + h * 16;
    float wgt[16];
    float mx = -3.4e38f;
#pragma unroll
    for (int p = 0; p < 16; ++p) { wgt[p] = attn_raw[at_base + p]; mx = fmaxf(mx, wgt[p]); }
    float s = 0.0f;
#pragma unroll
    for (int p = 0; p < 16; ++p) { wgt[p] = __expf(wgt[p] - mx); s += wgt[p]; }
    const float inv = 1.0f / s;

    const size_t off_base = (size_t)t * D_MODEL + h * (N_LEVELS * N_POINTS * 2);
    const float* rp = ref + (size_t)t * (N_LEVELS * 2);

    float acc = 0.0f;
#pragma unroll
    for (int lvl = 0; lvl < N_LEVELS; ++lvl) {
        const int Hl = Hs[lvl], Wl = Ws[lvl], st = starts[lvl];
        const float rx = rp[lvl * 2 + 0];
        const float ry = rp[lvl * 2 + 1];
#pragma unroll
        for (int p = 0; p < N_POINTS; ++p) {
            const float ox = off_raw[off_base + lvl * 8 + p * 2 + 0];
            const float oy = off_raw[off_base + lvl * 8 + p * 2 + 1];
            const float lx = rx + ox / (float)Wl;
            const float ly = ry + oy / (float)Hl;
            const float x  = lx * (float)Wl - 0.5f;
            const float y  = ly * (float)Hl - 0.5f;
            const float x0f = floorf(x), y0f = floorf(y);
            const float dx = x - x0f, dy = y - y0f;
            const int   x0 = (int)x0f, y0 = (int)y0f;
            const float aw = wgt[lvl * 4 + p] * inv;
            float sv = 0.0f;
            sv += ms_corner(value, b, st, Hl, Wl, x0,     y0,     (1.0f - dx) * (1.0f - dy), h, lane);
            sv += ms_corner(value, b, st, Hl, Wl, x0 + 1, y0,     dx * (1.0f - dy),          h, lane);
            sv += ms_corner(value, b, st, Hl, Wl, x0,     y0 + 1, (1.0f - dx) * dy,          h, lane);
            sv += ms_corner(value, b, st, Hl, Wl, x0 + 1, y0 + 1, dx * dy,                   h, lane);
            acc += aw * sv;
        }
    }
    out[(size_t)t * D_MODEL + h * D_HEAD + lane] = acc;
}

// ---------------------------------------------------------------------------
// out = LayerNorm(x0 + x1) * g + b, row width 256.
// One wave per row (32 lanes x 8 channels), 8 rows per block.
// ---------------------------------------------------------------------------
__global__ __launch_bounds__(256)
void add_layernorm(const float* __restrict__ x0,
                   const float* __restrict__ x1,
                   const float* __restrict__ gamma,
                   const float* __restrict__ beta,
                   float* __restrict__ out)
{
    const int lane = threadIdx.x & 31;
    const int wave = threadIdx.x >> 5;
    const int row  = blockIdx.x * 8 + wave;
    const size_t base = (size_t)row * D_MODEL;

    float v[8];
    float s = 0.0f;
#pragma unroll
    for (int i = 0; i < 8; ++i) {
        const int c = lane * 8 + i;
        v[i] = x0[base + c] + x1[base + c];
        s += v[i];
    }
#pragma unroll
    for (int off = 16; off >= 1; off >>= 1) s += __shfl_xor(s, off, 32);
    const float mean = s * (1.0f / D_MODEL);

    float vs = 0.0f;
#pragma unroll
    for (int i = 0; i < 8; ++i) { const float d = v[i] - mean; vs += d * d; }
#pragma unroll
    for (int off = 16; off >= 1; off >>= 1) vs += __shfl_xor(vs, off, 32);
    const float rstd = rsqrtf(vs * (1.0f / D_MODEL) + 1e-5f);

#pragma unroll
    for (int i = 0; i < 8; ++i) {
        const int c = lane * 8 + i;
        out[base + c] = (v[i] - mean) * rstd * gamma[c] + beta[c];
    }
}

// ---------------------------------------------------------------------------
// Launcher
// ---------------------------------------------------------------------------
extern "C" void kernel_launch(void* const* d_in, const int* in_sizes, int n_in,
                              void* d_out, int out_size, void* d_ws, size_t ws_size,
                              hipStream_t stream) {
    (void)in_sizes; (void)n_in; (void)out_size; (void)ws_size;

    const float* src   = (const float*)d_in[0];
    const float* pos   = (const float*)d_in[1];
    const float* ref   = (const float*)d_in[2];
    // d_in[3]=spatial_shapes, d_in[4]=level_start_index (hard-coded constants)
    const unsigned char* pmask = (const unsigned char*)d_in[5];
    const float* w_off = (const float*)d_in[6];
    const float* b_off = (const float*)d_in[7];
    const float* w_at  = (const float*)d_in[8];
    const float* b_at  = (const float*)d_in[9];
    const float* w_val = (const float*)d_in[10];
    const float* b_val = (const float*)d_in[11];
    const float* w_out = (const float*)d_in[12];
    const float* b_out = (const float*)d_in[13];
    const float* ln1_g = (const float*)d_in[14];
    const float* ln1_b = (const float*)d_in[15];
    const float* w1    = (const float*)d_in[16];
    const float* b1    = (const float*)d_in[17];
    const float* w2    = (const float*)d_in[18];
    const float* b2    = (const float*)d_in[19];
    const float* ln2_g = (const float*)d_in[20];
    const float* ln2_b = (const float*)d_in[21];

    float* outp = (float*)d_out;

    const size_t S1  = (size_t)M_TOK * D_MODEL * sizeof(float);   // 44.6 MB
    const size_t Sat = (size_t)M_TOK * 128 * sizeof(float);       // 22.3 MB
    char* ws = (char*)d_ws;
    float* ws_value = (float*)(ws);                    // value; later attn_out
    float* ws_off   = (float*)(ws + S1);               // offsets; later x (post-LN1)
    float* ws_attn  = (float*)(ws + 2 * S1);           // attn logits
    float* ws_samp  = (float*)(ws + 2 * S1 + Sat);     // sampled out; later ffn out
    float* ws_h1    = (float*)(ws + 3 * S1 + Sat);     // FFN hidden [M,1024]

    const dim3 blk(256, 1, 1);
    const int mBlocks = M_TOK / 64;                    // 680

    // 1) value = src @ w_val + b_val, masked rows zeroed
    gemm_wmma_f16<false><<<dim3(D_MODEL / 64, mBlocks), blk, 0, stream>>>(
        src, nullptr, w_val, b_val, ws_value, M_TOK, D_MODEL, D_MODEL, 0, pmask);
    // 2) off = (src+pos) @ w_off + b_off
    gemm_wmma_f16<true><<<dim3(D_MODEL / 64, mBlocks), blk, 0, stream>>>(
        src, pos, w_off, b_off, ws_off, M_TOK, D_MODEL, D_MODEL, 0, nullptr);
    // 3) attn logits = (src+pos) @ w_attn + b_attn
    gemm_wmma_f16<true><<<dim3(128 / 64, mBlocks), blk, 0, stream>>>(
        src, pos, w_at, b_at, ws_attn, M_TOK, 128, D_MODEL, 0, nullptr);
    // 4) deformable sampling (softmax fused)
    msdeform_sample<<<dim3(M_TOK, 1, 1), blk, 0, stream>>>(
        ws_value, ws_off, ws_attn, ref, ws_samp);
    // 5) attn_out = samp @ w_out + b_out  (reuses value region)
    gemm_wmma_f16<false><<<dim3(D_MODEL / 64, mBlocks), blk, 0, stream>>>(
        ws_samp, nullptr, w_out, b_out, ws_value, M_TOK, D_MODEL, D_MODEL, 0, nullptr);
    // 6) x = LN(src + attn_out)  (writes into off region)
    add_layernorm<<<dim3(M_TOK / 8, 1, 1), blk, 0, stream>>>(
        src, ws_value, ln1_g, ln1_b, ws_off);
    // 7) h1 = relu(x @ w1 + b1)
    gemm_wmma_f16<false><<<dim3(D_FFN / 64, mBlocks), blk, 0, stream>>>(
        ws_off, nullptr, w1, b1, ws_h1, M_TOK, D_FFN, D_MODEL, 1, nullptr);
    // 8) ffn = h1 @ w2 + b2  (reuses samp region)
    gemm_wmma_f16<false><<<dim3(D_MODEL / 64, mBlocks), blk, 0, stream>>>(
        ws_h1, nullptr, w2, b2, ws_samp, M_TOK, D_MODEL, D_FFN, 0, nullptr);
    // 9) out = LN(x + ffn)
    add_layernorm<<<dim3(M_TOK / 8, 1, 1), blk, 0, stream>>>(
        ws_off, ws_samp, ln2_g, ln2_b, outp);
}
